// GatedGCNNet_6038724018832
// MI455X (gfx1250) — matrix operand order repaired and checked by reading, not compiled, not run on hardware
//
#include <hip/hip_runtime.h>
#include <hip/hip_bf16.h>

// ---------------- problem constants ----------------
#define BGRAPHS 128
#define N_PER   1024
#define NNODES  (BGRAPHS * N_PER)   // 131072
#define KNN     7
#define XSIZE   16
#define H1      64
#define H2      128
#define BN_EPS  1e-5f

typedef __attribute__((ext_vector_type(16))) _Float16     v16h;
typedef __attribute__((ext_vector_type(8)))  _Float16     v8h;
typedef __attribute__((ext_vector_type(8)))  float        v8f;
typedef __attribute__((ext_vector_type(4)))  unsigned int v4u;
typedef __attribute__((ext_vector_type(8)))  int          v8i;
typedef __attribute__((ext_vector_type(4)))  int          v4i;

// ---------------- kNN: one block per graph, TDM-staged pos tile in LDS ----------------
// The 1024x3 f32 position tile (12KB, contiguous) is DMA'd into LDS by the Tensor Data
// Mover (1-row tile of 3072 x 4B elements), issued by wave 0; completion via TENSORcnt.
__global__ void knn_kernel(const float* __restrict__ pos, int* __restrict__ nbr) {
    __shared__ float p[N_PER * 3];               // AoS: x,y,z per node (memory order)
    const int g    = blockIdx.x;
    const int base = g * N_PER;

    if (threadIdx.x < 32) {                      // wave 0 issues the TDM load
        const unsigned long long ga =
            (unsigned long long)(uintptr_t)(pos + (size_t)base * 3);
        const unsigned int lds = (unsigned int)(uintptr_t)(void*)p;
        // D# group 0: count=1 | lds_addr | global_addr(57b) | type=2 ("image")
        v4u g0 = { 1u,
                   lds,
                   (unsigned int)(ga & 0xffffffffu),
                   (unsigned int)((ga >> 32) & 0x01ffffffu) | (2u << 30) };
        // D# group 1: data_size=4B; tensor_dim0=3072; tensor_dim1=1; tile_dim0=3072;
        //             tile_dim1=1; tensor_dim0_stride=3072
        v8i g1 = { (int)(2u << 16),              // [17:16] data_size = 2 (4 bytes)
                   (int)(3072u << 16),           // [63:48] tensor_dim0[15:0]
                   (int)(1u << 16),              // [79:64] dim0[31:16]=0 | [95:80] tensor_dim1[15:0]=1
                   (int)(3072u << 16),           // [111:96] dim1[31:16]=0 | [127:112] tile_dim0
                   1,                            // [143:128] tile_dim1=1 | [159:144] tile_dim2=0
                   3072,                         // [191:160] tensor_dim0_stride[31:0]
                   0, 0 };
        v4i gz4 = { 0, 0, 0, 0 };                // groups 2/3 unused (<=2D tensor)
        v8i gz8 = { 0, 0, 0, 0, 0, 0, 0, 0 };    // extra group (6-arg toolchain form)
        __builtin_amdgcn_tensor_load_to_lds(g0, g1, gz4, gz4, gz8, 0);
        __builtin_amdgcn_s_wait_tensorcnt(0);
    }
    __syncthreads();

    for (int i = threadIdx.x; i < N_PER; i += blockDim.x) {
        const float xi = p[i * 3 + 0], yi = p[i * 3 + 1], zi = p[i * 3 + 2];
        float bd[KNN]; int bi[KNN];
#pragma unroll
        for (int t = 0; t < KNN; ++t) { bd[t] = 3.0e38f; bi[t] = 0; }
        for (int j = 0; j < N_PER; ++j) {        // j is wave-uniform: LDS broadcasts
            if (j == i) continue;
            const float dx = p[j * 3 + 0] - xi;
            const float dy = p[j * 3 + 1] - yi;
            const float dz = p[j * 3 + 2] - zi;
            const float d = dx * dx + dy * dy + dz * dz;
            if (d < bd[KNN - 1]) {
                bd[KNN - 1] = d; bi[KNN - 1] = j;
#pragma unroll
                for (int t = KNN - 1; t > 0; --t) {
                    if (bd[t] < bd[t - 1]) {
                        float td = bd[t]; bd[t] = bd[t - 1]; bd[t - 1] = td;
                        int   ti = bi[t]; bi[t] = bi[t - 1]; bi[t - 1] = ti;
                    }
                }
            }
        }
#pragma unroll
        for (int t = 0; t < KNN; ++t) nbr[(size_t)(base + i) * KNN + t] = base + bi[t];
    }
}

// ---------------- small helpers ----------------
__global__ void cvt_f32_to_f16(const float* __restrict__ in, _Float16* __restrict__ out, long n) {
    long i = (long)blockIdx.x * blockDim.x + threadIdx.x;
    if (i < n) out[i] = (_Float16)in[i];
}

__global__ void zero_f32(float* __restrict__ p, int n) {
    int i = blockIdx.x * blockDim.x + threadIdx.x;
    if (i < n) p[i] = 0.0f;
}

// Pack the 4 f32 [din,dout] weight matrices straight into WMMA B-fragment order:
//   Wp[((ct*KT + kt)*32 + lane)*16 + h] = Wfused[k][ct*16 + n]   (f16, zero-padded K)
//   with k = kt*32 + (h<8 ? (lane>>4)*8 + h : 16 + (lane>>4)*8 + h-8), n = lane&15, Wfused = [wk|wq|wv|ws] columns.
__global__ void pack_wfrag(const float* __restrict__ wk, const float* __restrict__ wq,
                           const float* __restrict__ wv, const float* __restrict__ ws,
                           const float* __restrict__ bk, const float* __restrict__ bq,
                           const float* __restrict__ bv, const float* __restrict__ bs,
                           _Float16* __restrict__ wp, float* __restrict__ bcat,
                           int din, int dout) {
    const int cout = 4 * dout;
    const int KT   = (din + 31) / 32;
    const int tot  = (cout / 16) * KT * 32 * 16;
    for (int idx = blockIdx.x * blockDim.x + threadIdx.x; idx < tot;
         idx += gridDim.x * blockDim.x) {
        const int h    = idx & 15;
        const int lane = (idx >> 4) & 31;
        const int rest = idx >> 9;
        const int kt   = rest % KT;
        const int ct   = rest / KT;
        const int k    = kt * 32 + (h < 8 ? (lane >> 4) * 8 + h : 16 + (lane >> 4) * 8 + (h - 8));
        const int col  = ct * 16 + (lane & 15);
        const int grp  = col / dout, cc = col % dout;
        const float* w = (grp == 0) ? wk : (grp == 1) ? wq : (grp == 2) ? wv : ws;
        wp[idx] = (k < din) ? (_Float16)w[k * dout + cc] : (_Float16)0.0f;
        if (idx < cout) {
            const int bg = idx / dout;
            const float* b = (bg == 0) ? bk : (bg == 1) ? bq : (bg == 2) ? bv : bs;
            bcat[idx] = b[idx % dout];
        }
    }
}

// ---------------- WMMA GEMM: out[n,COUT] = A[n,KDIM](f16) @ Wfused + bias ----------------
// One wave computes a 64x16 output slab: 4 row tiles x 1 col tile, reusing B fragments.
// All A fragments are loaded up-front (clause of b128 loads) so the WMMA chain issues
// behind a single in-flight load window instead of load->wait->wmma serialization.
template <int KDIM, int COUT>
__global__ void gemm_wmma(const _Float16* __restrict__ A, const _Float16* __restrict__ Wp,
                          const float* __restrict__ bias, float* __restrict__ out, int n) {
    constexpr int CT = COUT / 16;
    constexpr int KT = (KDIM + 31) / 32;
    const int lane = threadIdx.x & 31;
    const int wv   = threadIdx.x >> 5;
    const long wid = (long)blockIdx.x * (blockDim.x >> 5) + wv;
    const long rowblk = wid / CT;                 // 64-row slab index
    const int  ct     = (int)(wid % CT);
    const long row0   = rowblk * 64;
    if (row0 >= n) return;                        // uniform per wave: EXEC stays all-ones
    const int m  = lane & 15;
    const int hi = lane >> 4;

    // B fragments (tiny, L2-resident, reused by every row slab)
    const v16h* wp16 = (const v16h*)Wp;
    v16h bfrag[KT];
#pragma unroll
    for (int kt = 0; kt < KT; ++kt) bfrag[kt] = wp16[(ct * KT + kt) * 32 + lane];

    // A fragments for all 4 row tiles
    v16h afrag[4 * KT];
#pragma unroll
    for (int mt = 0; mt < 4; ++mt) {
        const _Float16* Arow = A + (row0 + mt * 16 + m) * (long)KDIM;
#pragma unroll
        for (int kt = 0; kt < KT; ++kt) {
            const int kk = kt * 32;
            v8h alo = *(const v8h*)(Arow + kk + hi * 8);
            v8h ahi = {};
            if (kk + 16 < KDIM) ahi = *(const v8h*)(Arow + kk + 16 + hi * 8);
            afrag[mt * KT + kt] = __builtin_shufflevector(
                alo, ahi, 0, 1, 2, 3, 4, 5, 6, 7, 8, 9, 10, 11, 12, 13, 14, 15);
        }
    }

    v8f acc[4] = {};
#pragma unroll
    for (int mt = 0; mt < 4; ++mt)
#pragma unroll
        for (int kt = 0; kt < KT; ++kt)
            acc[mt] = __builtin_amdgcn_wmma_f32_16x16x32_f16(
                false, afrag[mt * KT + kt], false, bfrag[kt], (short)0, acc[mt],
                false, false);

    const float bb = bias[ct * 16 + m];
#pragma unroll
    for (int mt = 0; mt < 4; ++mt) {
        float* op = out + (row0 + mt * 16 + hi * 8) * (long)COUT + ct * 16 + m;
#pragma unroll
        for (int r = 0; r < 8; ++r) op[r * COUT] = acc[mt][r] + bb;
    }
}

// ---------------- gated message passing (dense: exactly KNN in-edges per dst) ------------
// buf layout per node i (float4 units, HQ = H/4): [ kf(HQ) | qf(HQ) | vf(HQ) | sf(HQ) ]
// out[i][c] = sf[i][c] + sum_t sigmoid(kf[i][c] + qf[j][c]) * vf[j][c],  j = nbr[i*K+t]
__global__ void msgpass4(const float4* __restrict__ buf, const int* __restrict__ nbr,
                         float4* __restrict__ out, long n, int HQ, int hqshift) {
    const long idx = (long)blockIdx.x * blockDim.x + threadIdx.x;
    const long tot = n << hqshift;
    if (idx >= tot) return;
    const int  c = (int)(idx & (HQ - 1));
    const long i = idx >> hqshift;
    const long stride = 4L << hqshift;           // 4*HQ float4s per node
    const float4 kf = buf[i * stride + c];
    float4 s = buf[i * stride + 3L * HQ + c];
#pragma unroll
    for (int t = 0; t < KNN; ++t) {
        const long j = nbr[i * KNN + t];
        const float4 q = buf[j * stride + HQ + c];
        const float4 v = buf[j * stride + 2L * HQ + c];
        s.x += v.x / (1.0f + __expf(-(kf.x + q.x)));
        s.y += v.y / (1.0f + __expf(-(kf.y + q.y)));
        s.z += v.z / (1.0f + __expf(-(kf.z + q.z)));
        s.w += v.w / (1.0f + __expf(-(kf.w + q.w)));
    }
    out[idx] = s;
}

// ---------------- batch-norm (training mode, biased var) ----------------
__global__ void bn_stats(const float* __restrict__ in, float* __restrict__ stats,
                         long n, int H) {
    extern __shared__ float sm[];                // [2*H]
    float* ssum = sm;
    float* ssq  = sm + H;
    for (int c = threadIdx.x; c < 2 * H; c += blockDim.x) sm[c] = 0.0f;
    __syncthreads();
    const long tot = n * H;
    for (long idx = (long)blockIdx.x * blockDim.x + threadIdx.x; idx < tot;
         idx += (long)gridDim.x * blockDim.x) {
        const float v = in[idx];
        const int c = (int)(idx & (H - 1));
        atomicAdd(&ssum[c], v);
        atomicAdd(&ssq[c], v * v);
    }
    __syncthreads();
    for (int c = threadIdx.x; c < H; c += blockDim.x) {
        atomicAdd(&stats[c], ssum[c]);
        atomicAdd(&stats[H + c], ssq[c]);
    }
}

template <typename OUT>
__global__ void bn_apply_relu(const float* __restrict__ in, const float* __restrict__ stats,
                              const float* __restrict__ gamma, const float* __restrict__ beta,
                              OUT* __restrict__ out, long n, int H) {
    const long tot = n * H;
    const long idx = (long)blockIdx.x * blockDim.x + threadIdx.x;
    if (idx >= tot) return;
    const int c = (int)(idx & (H - 1));
    const float inv_n = 1.0f / (float)n;
    const float mean = stats[c] * inv_n;
    const float var  = stats[H + c] * inv_n - mean * mean;
    const float rs   = rsqrtf(var + BN_EPS);
    float v = (in[idx] - mean) * rs * gamma[c] + beta[c];
    v = v > 0.0f ? v : 0.0f;
    out[idx] = (OUT)v;
}

// ---------------- per-graph mean pooling ----------------
__global__ void pool_mean(const float* __restrict__ h, float* __restrict__ out, int H) {
    const int g = blockIdx.x;          // graph
    const int c = threadIdx.x;         // channel (blockDim == H)
    float s = 0.0f;
    const float* base = h + (long)g * N_PER * H + c;
    for (int i = 0; i < N_PER; ++i) s += base[(long)i * H];
    out[(long)g * H + c] = s * (1.0f / (float)N_PER);
}

// ---------------- launch ----------------
extern "C" void kernel_launch(void* const* d_in, const int* in_sizes, int n_in,
                              void* d_out, int out_size, void* d_ws, size_t ws_size,
                              hipStream_t stream) {
    (void)in_sizes; (void)n_in; (void)out_size; (void)ws_size;
    const float* x   = (const float*)d_in[0];
    const float* pos = (const float*)d_in[1];
    // d_in[2] = batch (unused: graphs are contiguous equal-size blocks)
    const float* wk1 = (const float*)d_in[3];  const float* bk1 = (const float*)d_in[4];
    const float* wq1 = (const float*)d_in[5];  const float* bq1 = (const float*)d_in[6];
    const float* wv1 = (const float*)d_in[7];  const float* bv1 = (const float*)d_in[8];
    const float* ws1 = (const float*)d_in[9];  const float* bs1 = (const float*)d_in[10];
    const float* g1  = (const float*)d_in[11]; const float* be1 = (const float*)d_in[12];
    const float* wk2 = (const float*)d_in[13]; const float* bk2 = (const float*)d_in[14];
    const float* wq2 = (const float*)d_in[15]; const float* bq2 = (const float*)d_in[16];
    const float* wv2 = (const float*)d_in[17]; const float* bv2 = (const float*)d_in[18];
    const float* ws2 = (const float*)d_in[19]; const float* bs2 = (const float*)d_in[20];
    const float* g2  = (const float*)d_in[21]; const float* be2 = (const float*)d_in[22];

    // ----- workspace carve-up (256B aligned regions) -----
    char* ws = (char*)d_ws;
    size_t off = 0;
    auto carve = [&](size_t bytes) -> char* {
        char* p = ws + off;
        off += (bytes + 255) & ~(size_t)255;
        return p;
    };
    int*       nbr   = (int*)      carve((size_t)NNODES * KNN * sizeof(int));
    _Float16*  xh    = (_Float16*) carve((size_t)NNODES * XSIZE * sizeof(_Float16));
    _Float16*  wp1   = (_Float16*) carve((size_t)(4 * H1 / 16) * 1 * 32 * 16 * sizeof(_Float16));
    float*     b1cat = (float*)    carve((size_t)4 * H1 * sizeof(float));
    _Float16*  wp2   = (_Float16*) carve((size_t)(4 * H2 / 16) * 2 * 32 * 16 * sizeof(_Float16));
    float*     b2cat = (float*)    carve((size_t)4 * H2 * sizeof(float));
    float*     stats = (float*)    carve((size_t)(2 * H1 + 2 * H2) * sizeof(float));
    float*     buf   = (float*)    carve((size_t)NNODES * 4 * H2 * sizeof(float)); // shared L1/L2
    float*     conv  = (float*)    carve((size_t)NNODES * H2 * sizeof(float));     // shared L1/L2
    _Float16*  h1h   = (_Float16*) carve((size_t)NNODES * H1 * sizeof(_Float16));

    float* stats1 = stats;
    float* stats2 = stats + 2 * H1;

    // ----- pipeline -----
    zero_f32<<<2, 256, 0, stream>>>(stats, 2 * H1 + 2 * H2);

    knn_kernel<<<BGRAPHS, 256, 0, stream>>>(pos, nbr);

    cvt_f32_to_f16<<<(NNODES * XSIZE + 255) / 256, 256, 0, stream>>>(x, xh, (long)NNODES * XSIZE);

    pack_wfrag<<<32, 256, 0, stream>>>(wk1, wq1, wv1, ws1, bk1, bq1, bv1, bs1,
                                       wp1, b1cat, XSIZE, H1);
    pack_wfrag<<<128, 256, 0, stream>>>(wk2, wq2, wv2, ws2, bk2, bq2, bv2, bs2,
                                        wp2, b2cat, H1, H2);

    // ----- layer 1: fused GEMM -> gated message passing -> BN+ReLU (f16 out) -----
    {
        constexpr int COUT = 4 * H1;                              // 256
        const long waves = (long)(NNODES / 64) * (COUT / 16);     // 32768
        gemm_wmma<XSIZE, COUT><<<(int)(waves / 8), 256, 0, stream>>>(xh, wp1, b1cat, buf, NNODES);
        msgpass4<<<(NNODES * (H1 / 4) + 255) / 256, 256, 0, stream>>>(
            (const float4*)buf, nbr, (float4*)conv, (long)NNODES, H1 / 4, 4);
        bn_stats<<<512, 256, 2 * H1 * sizeof(float), stream>>>(conv, stats1, (long)NNODES, H1);
        bn_apply_relu<_Float16><<<(NNODES * H1 + 255) / 256, 256, 0, stream>>>(
            conv, stats1, g1, be1, h1h, (long)NNODES, H1);
    }

    // ----- layer 2 -----
    {
        constexpr int COUT = 4 * H2;                              // 512
        const long waves = (long)(NNODES / 64) * (COUT / 16);     // 65536
        gemm_wmma<H1, COUT><<<(int)(waves / 8), 256, 0, stream>>>(h1h, wp2, b2cat, buf, NNODES);
        msgpass4<<<(NNODES * (H2 / 4) + 255) / 256, 256, 0, stream>>>(
            (const float4*)buf, nbr, (float4*)conv, (long)NNODES, H2 / 4, 5);
        bn_stats<<<512, 256, 2 * H2 * sizeof(float), stream>>>(conv, stats2, (long)NNODES, H2);
        bn_apply_relu<float><<<(NNODES * H2 + 255) / 256, 256, 0, stream>>>(
            conv, stats2, g2, be2, conv, (long)NNODES, H2);       // normalize in place
    }

    // ----- per-graph mean pooling -----
    pool_mean<<<BGRAPHS, H2, 0, stream>>>(conv, (float*)d_out, H2);
}
